// LSTM_1743756722916
// MI455X (gfx1250) — matrix operand order repaired and checked by reading, not compiled, not run on hardware
//
#include <hip/hip_runtime.h>
#include <hip/hip_bf16.h>

// Problem constants (from reference):
#define NLAYERS 2
#define BATCH   64
#define SEQLEN  512
#define DH      512
#define KDIM    1024   // fan_in = D_H + D_IN

typedef __attribute__((ext_vector_type(16))) __bf16 v16bf;
typedef __attribute__((ext_vector_type(8)))  float  v8f;
typedef __attribute__((ext_vector_type(4)))  unsigned int v4u;

union Frag { v4u u[2]; v16bf h; };

__device__ __forceinline__ unsigned short f2bf(float f) {
    union { float f; unsigned int u; } v; v.f = f;
    unsigned int u = v.u;
    unsigned int r = u + 0x7FFFu + ((u >> 16) & 1u);   // round-to-nearest-even
    return (unsigned short)(r >> 16);
}

__device__ __forceinline__ float sigm(float x) {
    return 1.0f / (1.0f + __expf(-x));
}

// ---------------------------------------------------------------------------
// One-time: convert fp32 gate weights [L][K][N] -> bf16 packed column-major
// Wp[((l*4+g)*DH + n)*KDIM + k]  so each lane streams contiguous K for its n.
// ---------------------------------------------------------------------------
__global__ void pack_weights(const float* __restrict__ Wf,
                             const float* __restrict__ Wi,
                             const float* __restrict__ Wc,
                             const float* __restrict__ Wo,
                             unsigned short* __restrict__ Wp) {
    const size_t total = (size_t)NLAYERS * 4 * KDIM * DH;
    for (size_t e = (size_t)blockIdx.x * blockDim.x + threadIdx.x; e < total;
         e += (size_t)gridDim.x * blockDim.x) {
        int n = (int)(e % DH);
        size_t r = e / DH;
        int k = (int)(r % KDIM);
        r /= KDIM;
        int g = (int)(r % 4);
        int l = (int)(r / 4);
        const float* W = (g == 0) ? Wf : (g == 1) ? Wi : (g == 2) ? Wc : Wo;
        float v = W[((size_t)l * KDIM + k) * DH + n];
        Wp[(((size_t)l * 4 + g) * DH + n) * (size_t)KDIM + k] = f2bf(v);
    }
}

// ---------------------------------------------------------------------------
// One-time: zero cell state, seed hx buffers (parity 0) with h_init and seq[0]
// hxb: 4 buffers [layer][parity][BATCH][KDIM] bf16 (h in cols 0..511, x in 512..1023)
// ---------------------------------------------------------------------------
__global__ void init_state(const float* __restrict__ seq,
                           const float* __restrict__ h_init,
                           unsigned short* __restrict__ hxb,
                           float* __restrict__ Cst) {
    int tid = blockIdx.x * blockDim.x + threadIdx.x;
    int stride = gridDim.x * blockDim.x;
    for (int e = tid; e < NLAYERS * BATCH * DH; e += stride) Cst[e] = 0.0f;
    for (int e = tid; e < NLAYERS * BATCH * DH; e += stride) {
        int l = e / (BATCH * DH);
        int b = (e / DH) % BATCH;
        int n = e % DH;
        hxb[((size_t)(l * 2 + 0) * BATCH + b) * KDIM + n] = f2bf(h_init[e]);
    }
    for (int e = tid; e < BATCH * DH; e += stride) {
        int b = e / DH, n = e % DH;
        hxb[((size_t)(0 * 2 + 0) * BATCH + b) * KDIM + 512 + n] = f2bf(seq[e]);
    }
}

// ---------------------------------------------------------------------------
// Fused per-(step,layer) kernel: 4-gate GEMM [64x1024]x[1024x2048] via WMMA
// bf16->f32, then LSTM cell update.
// grid = 32 blocks x 256 threads (8 waves); block owns a 16-column N-slice.
// Each wave: one gate x 2 M-tiles x 1 N-tile; K-loop 32 -> 64 WMMA/wave with
// two independent accumulator chains (2-way ILP to cover WMMA latency).
// ---------------------------------------------------------------------------
__global__ __launch_bounds__(256)
void lstm_step(const unsigned short* __restrict__ hx,   // read buffer [64][1024] bf16
               const unsigned short* __restrict__ Wp,   // packed weights
               int gslot_base,                          // layer*4
               const float* __restrict__ bF, const float* __restrict__ bI,
               const float* __restrict__ bC, const float* __restrict__ bO,
               float* __restrict__ Cst,                 // [64][512] cell state
               unsigned short* __restrict__ h_self,     // write buffer (parity^1), h part
               unsigned short* __restrict__ h_feed,     // layer0: hx1[p]+512, else null
               float* __restrict__ h_out,               // layer1: d_out + t*B*D, else null
               float* __restrict__ h_final,             // last step: final-h slot, else null
               const float* __restrict__ seq_next,      // layer1 & t+1<S: seq[t+1], else null
               unsigned short* __restrict__ hx0x_next)  // hx0[p^1]+512 (x slot for next step)
{
    __shared__ float gbuf[4 * 64 * 16];   // 16 KB: [gate][m(batch)][n-local]

    const int tid   = threadIdx.x;
    const int wave  = tid >> 5;
    const int lane  = tid & 31;
    const int gate  = wave & 3;     // 0=f 1=i 2=c 3=o
    const int mpair = wave >> 2;    // 0 -> rows 0..31, 1 -> rows 32..63
    const int nbase = blockIdx.x * 16;
    const int laneN = lane & 15;
    const int hi    = lane >> 4;

    v8f acc0 = (v8f)(0.0f);
    v8f acc1 = (v8f)(0.0f);

    // A: row-major bf16, lane holds row mtile*16 + laneN; K chunks per 16-bit A layout
    const unsigned short* Arow0 = hx + (size_t)(mpair * 32 + laneN) * KDIM;
    const unsigned short* Arow1 = Arow0 + (size_t)16 * KDIM;
    // B: column-major packed, lane holds column nbase + laneN for this gate
    const unsigned short* Bcol =
        Wp + ((size_t)(gslot_base + gate) * DH + nbase + laneN) * (size_t)KDIM;

#pragma unroll 4
    for (int kt = 0; kt < KDIM / 32; ++kt) {
        const int ka = kt * 32 + hi * 8;     // A: lanes<16 K 0..7 & 16..23; hi lanes +8
        Frag a0, a1;
        a0.u[0] = *(const v4u*)(Arow0 + ka);
        a0.u[1] = *(const v4u*)(Arow0 + ka + 16);
        a1.u[0] = *(const v4u*)(Arow1 + ka);
        a1.u[1] = *(const v4u*)(Arow1 + ka + 16);

        const int kb = kt * 32 + hi * 8;     // B: lanes<16 K 0..7 & 16..23; hi lanes +8
        Frag b;
        b.u[0] = *(const v4u*)(Bcol + kb);
        b.u[1] = *(const v4u*)(Bcol + kb + 16);

        acc0 = __builtin_amdgcn_wmma_f32_16x16x32_bf16(
            false, a0.h, false, b.h, (short)0, acc0, false, false);
        acc1 = __builtin_amdgcn_wmma_f32_16x16x32_bf16(
            false, a1.h, false, b.h, (short)0, acc1, false, false);
    }

    // Scatter accumulators to LDS: C/D layout -> VGPR r, lanes0-15: M=r, lanes16-31: M=8+r
#pragma unroll
    for (int r = 0; r < 8; ++r) {
        int mrow0 = mpair * 32 + hi * 8 + r;
        int mrow1 = mrow0 + 16;
        gbuf[(gate * 64 + mrow0) * 16 + laneN] = acc0[r];
        gbuf[(gate * 64 + mrow1) * 16 + laneN] = acc1[r];
    }
    __syncthreads();

    // Fused LSTM cell update over this block's 64x16 slice (4 elems/thread)
    for (int e = tid; e < 64 * 16; e += 256) {
        int b  = e >> 4;
        int nl = e & 15;
        int n  = nbase + nl;
        float fpre = gbuf[(0 * 64 + b) * 16 + nl] + bF[n];
        float ipre = gbuf[(1 * 64 + b) * 16 + nl] + bI[n];
        float gpre = gbuf[(2 * 64 + b) * 16 + nl] + bC[n];
        float opre = gbuf[(3 * 64 + b) * 16 + nl] + bO[n];
        float fg = sigm(fpre);
        float ig = sigm(ipre);
        float gg = tanhf(gpre);
        float og = sigm(opre);
        float cold = Cst[b * DH + n];
        float cnew = fg * cold + ig * gg;
        float hv   = og * tanhf(cnew);
        Cst[b * DH + n] = cnew;
        unsigned short hb = f2bf(hv);
        h_self[(size_t)b * KDIM + n] = hb;                    // next-step recurrent input
        if (h_feed)  h_feed[(size_t)b * KDIM + n] = hb;       // layer-1 input this step
        if (h_out)   h_out[(size_t)b * DH + n] = hv;          // per-step last-layer output
        if (h_final) h_final[(size_t)b * DH + n] = hv;        // final-step h for this layer
    }

    // Layer-1 kernel also stages seq[t+1] -> bf16 x-slot of layer-0 next-parity buffer
    if (seq_next) {
        for (int e = blockIdx.x * 256 + tid; e < BATCH * DH; e += gridDim.x * 256) {
            int b = e / DH, n = e % DH;
            hx0x_next[(size_t)b * KDIM + n] = f2bf(seq_next[e]);
        }
    }
}

// ---------------------------------------------------------------------------
// Workspace layout (assumes ws_size >= ~9 MB):
//   [0, 8MB)            packed bf16 weights  (4.19M ushorts)
//   [8MB, 8MB+512KB)    4 hx ping-pong buffers [layer][parity][64][1024] bf16
//   [+512KB, +768KB)    cell state [2][64][512] f32
// ---------------------------------------------------------------------------
extern "C" void kernel_launch(void* const* d_in, const int* in_sizes, int n_in,
                              void* d_out, int out_size, void* d_ws, size_t ws_size,
                              hipStream_t stream) {
    (void)in_sizes; (void)n_in; (void)out_size; (void)ws_size;

    const float* seq    = (const float*)d_in[0];
    const float* h_init = (const float*)d_in[1];
    const float* Wf     = (const float*)d_in[2];
    const float* bFp    = (const float*)d_in[3];
    const float* Wi     = (const float*)d_in[4];
    const float* bIp    = (const float*)d_in[5];
    const float* Wc     = (const float*)d_in[6];
    const float* bCp    = (const float*)d_in[7];
    const float* Wo     = (const float*)d_in[8];
    const float* bOp    = (const float*)d_in[9];
    float* out = (float*)d_out;

    unsigned char* ws = (unsigned char*)d_ws;
    unsigned short* Wpack = (unsigned short*)ws;                              // 8 MB
    unsigned short* hxb   = (unsigned short*)(ws + (size_t)8 * 1024 * 1024);  // 4 x 128 KB
    float*          Cst   = (float*)(ws + (size_t)8 * 1024 * 1024
                                        + (size_t)4 * BATCH * KDIM * 2);      // 256 KB

    pack_weights<<<2048, 256, 0, stream>>>(Wf, Wi, Wc, Wo, Wpack);
    init_state<<<64, 256, 0, stream>>>(seq, h_init, hxb, Cst);

    const size_t hxBufElems = (size_t)BATCH * KDIM;
    for (int t = 0; t < SEQLEN; ++t) {
        const int p = t & 1;
        for (int l = 0; l < NLAYERS; ++l) {
            unsigned short* hx_rd = hxb + (size_t)(l * 2 + p)        * hxBufElems;
            unsigned short* hx_wr = hxb + (size_t)(l * 2 + (p ^ 1))  * hxBufElems;
            unsigned short* h_feed = (l == 0)
                ? (hxb + (size_t)(1 * 2 + p) * hxBufElems + 512) : nullptr;
            float* h_out = (l == 1) ? (out + (size_t)t * BATCH * DH) : nullptr;
            float* h_final = (t == SEQLEN - 1)
                ? (out + (size_t)SEQLEN * BATCH * DH + (size_t)l * BATCH * DH) : nullptr;
            const float* seq_next = (l == 1 && t + 1 < SEQLEN)
                ? (seq + (size_t)(t + 1) * BATCH * DH) : nullptr;
            unsigned short* hx0x_next =
                hxb + (size_t)(0 * 2 + ((t + 1) & 1)) * hxBufElems + 512;

            lstm_step<<<32, 256, 0, stream>>>(
                hx_rd, Wpack, l * 4,
                bFp + (size_t)l * DH, bIp + (size_t)l * DH,
                bCp + (size_t)l * DH, bOp + (size_t)l * DH,
                Cst + (size_t)l * BATCH * DH,
                hx_wr, h_feed, h_out, h_final, seq_next, hx0x_next);
        }
    }
}
